// MultiHeadSelfAttention_22582938043046
// MI455X (gfx1250) — matrix operand order repaired and checked by reading, not compiled, not run on hardware
//
#include <hip/hip_runtime.h>

typedef __attribute__((ext_vector_type(16))) __bf16 v16bf;
typedef __attribute__((ext_vector_type(8)))  float  v8f;
typedef __attribute__((ext_vector_type(4)))  int    v4i;   // native vector (not HIP_vector_type)

#define D_MODEL 1024
#define NUM_HEADS 16
#define D_K 64
#define SEQ 64
#define BATCH 512
#define M_TOTAL (BATCH * SEQ)

// Async global->LDS copy (CDNA5): guarded so compilation survives toolchains
// that lack the builtin; fallback is a register-staged copy.
#if defined(__has_builtin)
#  if __has_builtin(__builtin_amdgcn_global_load_async_to_lds_b128) && \
      __has_builtin(__builtin_amdgcn_s_wait_asynccnt)
#    define ATTN_ASYNC 1
#  endif
#endif
#ifndef ATTN_ASYNC
#  define ATTN_ASYNC 0
#endif

// Builtin parameter types (from hipcc diagnostics): native int4 in AS1 / AS3.
typedef __attribute__((address_space(1))) v4i* g_v4p;
typedef __attribute__((address_space(3))) v4i* l_v4p;

__device__ __forceinline__ unsigned short f2bf(float f) {
  union { float f; unsigned u; } c; c.f = f;
  unsigned u = c.u;
  return (unsigned short)((u + 0x7FFFu + ((u >> 16) & 1u)) >> 16);
}

// A-fragment (16x32 bf16, row-major tile in LDS):
// lanes 0-15: row M=lane, K = {0..7, 16..23}; lanes 16-31: row M=lane-16, K = {8..15, 24..31}
__device__ __forceinline__ v16bf load_fragA(const unsigned short* tile, int row, int stride, int lane) {
  const unsigned short* p = tile + row * stride + ((lane & 16) ? 8 : 0);
  union { uint4 u[2]; v16bf v; } f;
  f.u[0] = *(const uint4*)(p);
  f.u[1] = *(const uint4*)(p + 16);
  return f.v;
}

// B-fragment (32x16 bf16; tile stored as N rows x K cols, i.e. B^T row-major):
// lanes 0-15: col N=lane, K = {0..7, 16..23}; lanes 16-31: col N=lane-16, K = {8..15, 24..31}
__device__ __forceinline__ v16bf load_fragB(const unsigned short* tile, int row, int stride, int lane) {
  const unsigned short* p = tile + row * stride + ((lane & 16) ? 8 : 0);
  union { uint4 u[2]; v16bf v; } f;
  f.u[0] = *(const uint4*)(p);
  f.u[1] = *(const uint4*)(p + 16);
  return f.v;
}

// ---------------------------------------------------------------------------
// Y[M,1024] = X[M,1024] @ W[1024,1024]^T + bias ; bf16 WMMA, f32 accum.
// XBF: X is bf16 (straight 16B copies to LDS) vs fp32 (convert on stage).
// BHSD==1: Y = bf16 in [B,H,S,Dk]; BHSD==0: Y = fp32 row-major [M,N].
// Register-staged software pipeline: global fetch of slice kk+32 overlaps the
// WMMAs consuming LDS slice kk.
// ---------------------------------------------------------------------------
#define GK   D_MODEL
#define GN   D_MODEL
#define GT_M 128
#define GT_N 128
#define GT_K 32
#define GLDA 40        // LDS row stride in bf16 elements (80B rows, 16B aligned)

template <int BHSD, bool XBF>
__global__ __launch_bounds__(256) void gemm_xwT(
    const void* __restrict__ Xv, const float* __restrict__ W,
    const float* __restrict__ bias, void* __restrict__ Yv) {
  __shared__ unsigned short As[GT_M * GLDA];
  __shared__ unsigned short Bs[GT_N * GLDA];

  const int t    = threadIdx.x;
  const int lane = t & 31;
  const int wave = t >> 5;
  const int wm   = wave & 3;    // 4 m-groups of 32 rows
  const int wn   = wave >> 2;   // 2 n-groups of 64 cols
  const int m0   = blockIdx.y * GT_M;
  const int n0   = blockIdx.x * GT_N;

  // B (weights, fp32) geometry: 8 float4-cols x 32 rows
  const int c4 = t & 7;
  const int r  = t >> 3;        // 0..31
  const float* wp = W + (size_t)(n0 + r) * GK + c4 * 4;

  // A geometry (bf16 path): 4 x (8 bf16) cols x 64 rows
  const int c8 = t & 3;
  const int r8 = t >> 2;        // 0..63

  const float*          xf = (const float*)Xv          + (size_t)(m0 + r)  * GK + c4 * 4;
  const unsigned short* xb = (const unsigned short*)Xv + (size_t)(m0 + r8) * GK + c8 * 8;

  v8f acc[2][4] = {};
  float4 raf[4], rb[4];
  uint4  rab[2];

  // prologue fetch (kk = 0)
  #pragma unroll
  for (int j = 0; j < 4; ++j)
    rb[j] = *(const float4*)(wp + (size_t)(j * 32) * GK);
  if (XBF) {
    #pragma unroll
    for (int j = 0; j < 2; ++j)
      rab[j] = *(const uint4*)(xb + (size_t)(j * 64) * GK);
  } else {
    #pragma unroll
    for (int j = 0; j < 4; ++j)
      raf[j] = *(const float4*)(xf + (size_t)(j * 32) * GK);
  }

  for (int kk = 0; kk < GK; kk += GT_K) {
    // stage current slice regs -> LDS
    #pragma unroll
    for (int j = 0; j < 4; ++j) {
      const int row = r + j * 32;
      unsigned short* db = Bs + row * GLDA + c4 * 4;
      uint2 pb;
      pb.x = (unsigned)f2bf(rb[j].x) | ((unsigned)f2bf(rb[j].y) << 16);
      pb.y = (unsigned)f2bf(rb[j].z) | ((unsigned)f2bf(rb[j].w) << 16);
      *(uint2*)db = pb;
    }
    if (XBF) {
      #pragma unroll
      for (int j = 0; j < 2; ++j)
        *(uint4*)(As + (r8 + j * 64) * GLDA + c8 * 8) = rab[j];
    } else {
      #pragma unroll
      for (int j = 0; j < 4; ++j) {
        unsigned short* da = As + (r + j * 32) * GLDA + c4 * 4;
        uint2 pa;
        pa.x = (unsigned)f2bf(raf[j].x) | ((unsigned)f2bf(raf[j].y) << 16);
        pa.y = (unsigned)f2bf(raf[j].z) | ((unsigned)f2bf(raf[j].w) << 16);
        *(uint2*)da = pa;
      }
    }
    __syncthreads();

    // prefetch next K-slice (clamped; branchless) while WMMAs run below
    const int kn = (kk + GT_K < GK) ? (kk + GT_K) : kk;
    #pragma unroll
    for (int j = 0; j < 4; ++j)
      rb[j] = *(const float4*)(wp + (size_t)(j * 32) * GK + kn);
    if (XBF) {
      #pragma unroll
      for (int j = 0; j < 2; ++j)
        rab[j] = *(const uint4*)(xb + (size_t)(j * 64) * GK + kn);
    } else {
      #pragma unroll
      for (int j = 0; j < 4; ++j)
        raf[j] = *(const float4*)(xf + (size_t)(j * 32) * GK + kn);
    }

    v16bf af[2], bfr[4];
    #pragma unroll
    for (int tm = 0; tm < 2; ++tm)
      af[tm] = load_fragA(As, wm * 32 + tm * 16 + (lane & 15), GLDA, lane);
    #pragma unroll
    for (int tn = 0; tn < 4; ++tn)
      bfr[tn] = load_fragB(Bs, wn * 64 + tn * 16 + (lane & 15), GLDA, lane);

    #pragma unroll
    for (int tm = 0; tm < 2; ++tm)
      #pragma unroll
      for (int tn = 0; tn < 4; ++tn)
        acc[tm][tn] = __builtin_amdgcn_wmma_f32_16x16x32_bf16(
            false, af[tm], false, bfr[tn], (short)0, acc[tm][tn], false, false);
    __syncthreads();
  }

  const int mbase = m0 + wm * 32;
  const int nbase = n0 + wn * 64;
  #pragma unroll
  for (int tm = 0; tm < 2; ++tm) {
    #pragma unroll
    for (int tn = 0; tn < 4; ++tn) {
      const int nn = nbase + tn * 16 + (lane & 15);
      const float bv = bias[nn];
      #pragma unroll
      for (int i = 0; i < 8; ++i) {
        const int mm = mbase + tm * 16 + i + ((lane & 16) ? 8 : 0);
        const float val = acc[tm][tn][i] + bv;
        if (BHSD) {
          // bf16 [B,H,S,Dk]: b = mm/64, s = mm%64, h = nn/64, dk = nn%64
          const int b = mm >> 6, s = mm & 63, hh = nn >> 6, dk = nn & 63;
          ((unsigned short*)Yv)[(((size_t)(b * NUM_HEADS + hh) * SEQ + s) << 6) + dk] = f2bf(val);
        } else {
          ((float*)Yv)[(size_t)mm * GN + nn] = val;
        }
      }
    }
  }
}

// ---------------------------------------------------------------------------
// Per-(b,h) attention on bf16 Q/K/V: scores = QK^T/8 + relpos bias, softmax,
// ctx = P V. One workgroup (8 waves) per head instance, all tiles in LDS.
// Q/K staged via async global->LDS copies when available.
// ---------------------------------------------------------------------------
#define ALD 72   // bf16 tile stride (144B rows, 16B aligned)
#define SLD 68   // fp32 score stride

__global__ __launch_bounds__(256) void attention_kernel(
    const unsigned short* __restrict__ Q, const unsigned short* __restrict__ K,
    const unsigned short* __restrict__ V, const float* __restrict__ bias_table,
    unsigned short* __restrict__ ctx) {
  __shared__ unsigned short Qs[SEQ * ALD];
  __shared__ unsigned short Ks[SEQ * ALD];
  __shared__ unsigned short Vt[D_K * ALD];   // Vt[dk][s] (transposed)
  __shared__ float          Sc[SEQ * SLD];
  __shared__ unsigned short Ps[SEQ * ALD];

  const int bh = blockIdx.x;
  const int b  = bh >> 4;
  const int h  = bh & 15;
  const size_t base = (size_t)bh * SEQ * D_K;

  const int t    = threadIdx.x;
  const int lane = t & 31;
  const int wave = t >> 5;
  const int qm   = wave & 3;    // 16-row q tile
  const int nh   = wave >> 2;   // 0/1 -> 32-col half

  {
    const int c8 = t & 7;       // 8 groups of 8 bf16 -> 64 cols
    const int r8 = t >> 3;      // 0..31
#if ATTN_ASYNC
    #pragma unroll
    for (int j = 0; j < 2; ++j) {
      const int row = r8 + j * 32;
      __builtin_amdgcn_global_load_async_to_lds_b128(
          (g_v4p)(Q + base + row * D_K + c8 * 8),
          (l_v4p)(Qs + row * ALD + c8 * 8), 0, 0);
      __builtin_amdgcn_global_load_async_to_lds_b128(
          (g_v4p)(K + base + row * D_K + c8 * 8),
          (l_v4p)(Ks + row * ALD + c8 * 8), 0, 0);
    }
#else
    #pragma unroll
    for (int j = 0; j < 2; ++j) {
      const int row = r8 + j * 32;
      *(uint4*)(Qs + row * ALD + c8 * 8) = *(const uint4*)(Q + base + row * D_K + c8 * 8);
      *(uint4*)(Ks + row * ALD + c8 * 8) = *(const uint4*)(K + base + row * D_K + c8 * 8);
    }
#endif
    // V: transposed into LDS (B-fragments of P@V need V columns contiguous)
    #pragma unroll
    for (int j = 0; j < 2; ++j) {
      const int row = r8 + j * 32;
      union { uint4 u; unsigned short s[8]; } v;
      v.u = *(const uint4*)(V + base + row * D_K + c8 * 8);
      #pragma unroll
      for (int e = 0; e < 8; ++e)
        Vt[(c8 * 8 + e) * ALD + row] = v.s[e];
    }
#if ATTN_ASYNC
    __builtin_amdgcn_s_wait_asynccnt(0);
#endif
  }
  __syncthreads();

  // scores = Q @ K^T  (each wave: 16 q-rows x 32 k-cols)
  {
    v8f acc[2] = {};
    #pragma unroll
    for (int kk = 0; kk < D_K; kk += 32) {
      v16bf aq = load_fragA(Qs + kk, qm * 16 + (lane & 15), ALD, lane);
      #pragma unroll
      for (int tn = 0; tn < 2; ++tn) {
        v16bf bk = load_fragB(Ks + kk, nh * 32 + tn * 16 + (lane & 15), ALD, lane);
        acc[tn] = __builtin_amdgcn_wmma_f32_16x16x32_bf16(
            false, aq, false, bk, (short)0, acc[tn], false, false);
      }
    }
    #pragma unroll
    for (int tn = 0; tn < 2; ++tn) {
      const int n = nh * 32 + tn * 16 + (lane & 15);
      #pragma unroll
      for (int i = 0; i < 8; ++i) {
        const int m = qm * 16 + i + ((lane & 16) ? 8 : 0);
        Sc[m * SLD + n] = acc[tn][i];
      }
    }
  }
  __syncthreads();

  // softmax with chess-board relative position bias (one row per thread)
  if (t < SEQ) {
    const int q  = t;
    const int fq = q & 7, rq = q >> 3;
    float mx = -3.4e38f;
    for (int k2 = 0; k2 < SEQ; ++k2) {
      const int fk = k2 & 7, rk = k2 >> 3;
      const int rf  = fq - fk + 7;   // always in [0,14] for an 8x8 board
      const int rr2 = rq - rk + 7;
      const float bb = bias_table[(rf * 15 + rr2) * NUM_HEADS + h];
      const float v = Sc[q * SLD + k2] * 0.125f + bb;
      mx = fmaxf(mx, v);
    }
    float sum = 0.f;
    for (int k2 = 0; k2 < SEQ; ++k2) {
      const int fk = k2 & 7, rk = k2 >> 3;
      const int rf  = fq - fk + 7;
      const int rr2 = rq - rk + 7;
      const float bb = bias_table[(rf * 15 + rr2) * NUM_HEADS + h];
      const float e = __expf(Sc[q * SLD + k2] * 0.125f + bb - mx);
      sum += e;
      Sc[q * SLD + k2] = e;
    }
    const float inv = 1.0f / sum;
    for (int k2 = 0; k2 < SEQ; ++k2)
      Ps[q * ALD + k2] = f2bf(Sc[q * SLD + k2] * inv);
  }
  __syncthreads();

  // ctx = P @ V   (B(k,n) = V[k][n] = Vt[n][k], contiguous in k)
  {
    v8f acc[2] = {};
    #pragma unroll
    for (int kk = 0; kk < SEQ; kk += 32) {
      v16bf ap = load_fragA(Ps + kk, qm * 16 + (lane & 15), ALD, lane);
      #pragma unroll
      for (int tn = 0; tn < 2; ++tn) {
        v16bf bv = load_fragB(Vt + kk, nh * 32 + tn * 16 + (lane & 15), ALD, lane);
        acc[tn] = __builtin_amdgcn_wmma_f32_16x16x32_bf16(
            false, ap, false, bv, (short)0, acc[tn], false, false);
      }
    }
    #pragma unroll
    for (int tn = 0; tn < 2; ++tn) {
      const int dk = nh * 32 + tn * 16 + (lane & 15);
      #pragma unroll
      for (int i = 0; i < 8; ++i) {
        const int m = qm * 16 + i + ((lane & 16) ? 8 : 0);
        ctx[((size_t)b * SEQ + m) * D_MODEL + h * D_K + dk] = f2bf(acc[tn][i]);
      }
    }
  }
}

extern "C" void kernel_launch(void* const* d_in, const int* in_sizes, int n_in,
                              void* d_out, int out_size, void* d_ws, size_t ws_size,
                              hipStream_t stream) {
  const float* x  = (const float*)d_in[0];
  const float* Wq = (const float*)d_in[1];
  const float* bq = (const float*)d_in[2];
  const float* Wk = (const float*)d_in[3];
  const float* bk = (const float*)d_in[4];
  const float* Wv = (const float*)d_in[5];
  const float* bv = (const float*)d_in[6];
  const float* Wo = (const float*)d_in[7];
  const float* bo = (const float*)d_in[8];
  const float* bt = (const float*)d_in[9];

  const size_t mat = (size_t)M_TOTAL * D_MODEL;   // elements
  unsigned short* Qw = (unsigned short*)d_ws;     // bf16 intermediates
  unsigned short* Kw = Qw + mat;
  unsigned short* Vw = Kw + mat;
  unsigned short* Cw = Vw + mat;

  dim3 blk(256);
  dim3 grd(GN / GT_N, M_TOTAL / GT_M);
  gemm_xwT<1, false><<<grd, blk, 0, stream>>>(x, Wq, bq, Qw);
  gemm_xwT<1, false><<<grd, blk, 0, stream>>>(x, Wk, bk, Kw);
  gemm_xwT<1, false><<<grd, blk, 0, stream>>>(x, Wv, bv, Vw);
  attention_kernel<<<dim3(BATCH * NUM_HEADS), blk, 0, stream>>>(Qw, Kw, Vw, bt, Cw);
  gemm_xwT<0, true><<<grd, blk, 0, stream>>>(Cw, Wo, bo, d_out);
}
